// WA_Conv2d_48962627175198
// MI455X (gfx1250) — matrix operand order repaired and checked by reading, not compiled, not run on hardware
//
#include <hip/hip_runtime.h>
#include <hip/hip_bf16.h>

typedef __attribute__((ext_vector_type(16))) _Float16     v16h;
typedef __attribute__((ext_vector_type(8)))  float        v8f;
typedef __attribute__((ext_vector_type(4)))  unsigned int v4u;
typedef __attribute__((ext_vector_type(4)))  int          v4i;
typedef __attribute__((ext_vector_type(8)))  int          v8i;

#define BB    32
#define CINC  64
#define COUTC 64
#define HH    112
#define WW    112
#define NPIX  (HH * WW)          // 12544
#define KTOT  (CINC * 9)         // 576
#define OLEN  (COUTC * CINC * 9) // 36864
#define TN    128                // output pixels per block (8 N-tiles)

// ---------------------------------------------------------------------------
// Kernel 1: global average pool.  pooled[b*CIN+ci] = mean(x[b,ci,:,:])
// ---------------------------------------------------------------------------
__global__ __launch_bounds__(256) void pool_kernel(const float* __restrict__ x,
                                                   float* __restrict__ pooled) {
  __shared__ float red[256];
  const int bc = blockIdx.x;                      // 0 .. B*CIN-1
  const float* p = x + (size_t)bc * NPIX;
  float s = 0.f;
  for (int i = threadIdx.x; i < NPIX; i += 256) s += p[i];
  red[threadIdx.x] = s;
  __syncthreads();
  for (int off = 128; off > 0; off >>= 1) {
    if ((int)threadIdx.x < off) red[threadIdx.x] += red[threadIdx.x + off];
    __syncthreads();
  }
  if (threadIdx.x == 0) pooled[bc] = red[0] * (1.0f / NPIX);
}

// ---------------------------------------------------------------------------
// Kernel 2: attention + dynamic-weight build.
//   att = sigmoid(pooled @ (wq+wk)^T);  dynw[b][o] = f16(weight[o] * att)
// ---------------------------------------------------------------------------
__global__ __launch_bounds__(256) void attn_kernel(const float* __restrict__ pooled,
                                                   const float* __restrict__ weight,
                                                   const float* __restrict__ wq,
                                                   const float* __restrict__ wk,
                                                   _Float16* __restrict__ dynw) {
  __shared__ float ws[64 * 64];        // (wq+wk) rows for this o-tile
  __shared__ float ps[BB * CINC];      // all pooled values (32x64)
  const int obase = blockIdx.x * 64;

  for (int i = threadIdx.x; i < 64 * 64; i += 256) {
    const size_t gi = (size_t)obase * 64 + i;
    ws[i] = wq[gi] + wk[gi];
  }
  for (int i = threadIdx.x; i < BB * CINC; i += 256) ps[i] = pooled[i];
  __syncthreads();

  const int ol = threadIdx.x & 63;
  const int b0 = threadIdx.x >> 6;
  const float wv = weight[obase + ol];
  const float* wr = &ws[ol * 64];

  for (int bj = 0; bj < 8; ++bj) {
    const int b = b0 * 8 + bj;
    const float* pr = &ps[b * CINC];
    float s = 0.f;
#pragma unroll 8
    for (int k = 0; k < 64; ++k) s += pr[k] * wr[k];
    const float att = 1.0f / (1.0f + __expf(-s));
    dynw[(size_t)b * OLEN + obase + ol] = (_Float16)(wv * att);
  }
}

// ---------------------------------------------------------------------------
// Kernel 3: per-sample implicit GEMM conv, WMMA f32 <- f16 x f16.
// Block = 128 thr (4 waves): one sample x 128 pixels x all 64 couts.
// Wave w: M-tile w (16 couts) x 8 N-tiles.  K-step = 32 (18 steps).
// A tile (dyn weights) is DMA'd into LDS by the Tensor Data Mover.
// ---------------------------------------------------------------------------
__global__ __launch_bounds__(128) void conv_wmma_kernel(const float* __restrict__ x,
                                                        const _Float16* __restrict__ dynw,
                                                        float* __restrict__ out) {
  __shared__ __align__(32) _Float16 Alds[COUTC * 32];  // [co][k]  64x32 f16 (4 KB)
  __shared__ __align__(32) _Float16 Blds[TN * 32];     // [n][k]  128x32 f16 (8 KB)

  const int b        = blockIdx.y;
  const int tileBase = blockIdx.x * TN;
  const int tid      = threadIdx.x;
  const int wave     = tid >> 5;
  const int lane     = tid & 31;
  const int l16      = lane & 15;
  const int hi       = lane >> 4;

  const float*    xb = x    + (size_t)b * CINC * NPIX;
  const _Float16* wb = dynw + (size_t)b * OLEN;

  // each thread owns one output-pixel column of the B tile
  const int n  = tid;
  const int p  = tileBase + n;
  const int oh = p / WW;
  const int ow = p - oh * WW;

  v8f acc[8];
#pragma unroll
  for (int j = 0; j < 8; ++j) acc[j] = (v8f){};

  for (int kb = 0; kb < KTOT; kb += 32) {
    // ---- A tile: 64 rows x 32 halfs of dynw[b][:, kb..kb+31] ----
#if __has_builtin(__builtin_amdgcn_tensor_load_to_lds)
    if (wave == 0) {
      // D# per CDNA5 ISA: 2D tile, data_size=2B,
      // tensor row pitch 576 elems, tile 32 (K) x 64 (couts).
      const unsigned lds_off =
          (unsigned)(size_t)(const void*)&Alds[0];         // LDS aperture: low 32 bits = offset
      const unsigned long long ga =
          (unsigned long long)(size_t)(const void*)(wb + kb);
      const v4u g0 = { 1u,                                  // count=1, user mode
                       lds_off,
                       (unsigned)ga,
                       (unsigned)((ga >> 32) & 0x1FFFFFFu) | 0x80000000u }; // type=2
      const v8i g1 = { 0x00010000,          // data_size=1 (2 bytes)
                       0x02400000,          // tensor_dim0 = 576 (bits 79:48 lo)
                       0x00400000,          // tensor_dim1 = 64
                       0x00200000,          // tile_dim0  = 32
                       64,                  // tile_dim1  = 64
                       576,                 // tensor_dim0_stride = 576
                       0, 0 };
      const v4i gz = { 0, 0, 0, 0 };
#if __clang_major__ >= 23
      const v8i gz8 = { 0, 0, 0, 0, 0, 0, 0, 0 };
      __builtin_amdgcn_tensor_load_to_lds(g0, g1, gz, gz, gz8, 0);
#else
      __builtin_amdgcn_tensor_load_to_lds(g0, g1, gz, gz, 0);
#endif
    }
#else
    for (int c = tid; c < 256; c += 128) {
      const int row = c >> 2;
      const int cb  = (c & 3) * 8;
      const uint4 v = *(const uint4*)(wb + (size_t)row * KTOT + kb + cb);
      *(uint4*)(&Alds[row * 32 + cb]) = v;
      if (kb + 32 < KTOT)
        __builtin_prefetch(wb + (size_t)row * KTOT + kb + 32 + cb, 0, 1);
    }
#endif

    // ---- B tile: im2col gather, Blds[n][k]; DMA overlaps this gather ----
#pragma unroll 4
    for (int k = 0; k < 32; ++k) {
      const int kg = kb + k;
      const int ci = kg / 9;
      const int r  = kg - ci * 9;
      const int kh = r / 3;
      const int kw = r - kh * 3;
      const int ih = oh + kh - 1;
      const int iw = ow + kw - 1;
      float v = 0.f;
      if ((unsigned)ih < (unsigned)HH && (unsigned)iw < (unsigned)WW)
        v = xb[(size_t)ci * NPIX + ih * WW + iw];
      Blds[n * 32 + k] = (_Float16)v;
    }

#if __has_builtin(__builtin_amdgcn_tensor_load_to_lds)
    if (wave == 0) {
#if __has_builtin(__builtin_amdgcn_s_wait_tensorcnt)
      __builtin_amdgcn_s_wait_tensorcnt(0);
#else
      asm volatile("s_wait_tensorcnt 0x0" ::: "memory");
#endif
    }
#endif
    __syncthreads();

    // ---- WMMA: A frag: lanes 0-15 row m, v0-7 = K hi*16..+15 (contiguous,
    //            matching B).  B frag: col l16, K = hi*16..+15.
    const int arow = wave * 16 + l16;
    union { uint4 u[2]; v16h h; } af;
    af.u[0] = *(const uint4*)(&Alds[arow * 32 + hi * 16]);
    af.u[1] = *(const uint4*)(&Alds[arow * 32 + hi * 16 + 8]);
    const v16h av = af.h;

#pragma unroll
    for (int j = 0; j < 8; ++j) {
      const v16h bv = *(const v16h*)(&Blds[(j * 16 + l16) * 32 + hi * 16]);
      acc[j] = __builtin_amdgcn_wmma_f32_16x16x32_f16(
          /*neg_a=*/false, av, /*neg_b=*/false, bv,
          /*c_mod=*/(short)0, acc[j], /*reuse_a=*/false, /*reuse_b=*/false);
    }
    __syncthreads();
  }

  // ---- epilogue: C layout = lane l16 -> col, VGPR r -> row hi*8+r ----
  const int coBase = wave * 16 + hi * 8;
#pragma unroll
  for (int j = 0; j < 8; ++j) {
    const int col = tileBase + j * 16 + l16;
#pragma unroll
    for (int r = 0; r < 8; ++r) {
      const int co = coBase + r;
      out[(size_t)(b * COUTC + co) * NPIX + col] = acc[j][r];
    }
  }
}

// ---------------------------------------------------------------------------
extern "C" void kernel_launch(void* const* d_in, const int* in_sizes, int n_in,
                              void* d_out, int out_size, void* d_ws, size_t ws_size,
                              hipStream_t stream) {
  (void)in_sizes; (void)n_in; (void)out_size; (void)ws_size;
  const float* x      = (const float*)d_in[0];
  const float* weight = (const float*)d_in[1];
  const float* wq     = (const float*)d_in[2];
  const float* wk     = (const float*)d_in[3];
  float* out = (float*)d_out;

  float*    pooled = (float*)d_ws;                       // 2048 f32 (8 KB)
  _Float16* dynw   = (_Float16*)((char*)d_ws + 8192);    // 32*36864 f16 (2.36 MB)

  pool_kernel<<<dim3(BB * CINC), 256, 0, stream>>>(x, pooled);
  attn_kernel<<<dim3(OLEN / 64), 256, 0, stream>>>(pooled, weight, wq, wk, dynw);
  conv_wmma_kernel<<<dim3(NPIX / TN, BB), 128, 0, stream>>>(x, dynw, out);
}
